// StochasticWavenet_11038065951121
// MI455X (gfx1250) — matrix operand order repaired and checked by reading, not compiled, not run on hardware
//
// Stochastic WaveNet on MI455X (gfx1250): bf16 WMMA GEMM formulation.
//
// Every conv1d is a channel GEMM (N = B*T = 65536). K=2 dilated convs are two
// shifted GEMM taps. ~150 DxD GEMMs -> ~320 GFLOP; activation traffic ~3-4 GB
// -> both sides of the roofline land ~0.1-0.3 ms, so we use
// v_wmma_f32_16x16x32_bf16 with f32 accumulation, LDS-transposed X tiles, and
// fused epilogues (bias/relu/clamp/skip-accumulate/tanh-transpose).
// Wave tile is 64(M)x32(N): 8 WMMA per k-step vs 12 b128 loads (1.5 loads/WMMA)
// for denser XDL issue than the 32x32 variant.
#include <hip/hip_runtime.h>
#include <hip/hip_bf16.h>
#include <math.h>

#define B_    16
#define T_    4096
#define DIN_  64
#define D_    128
#define DOUT_ 64
#define NSL_  4

typedef __attribute__((ext_vector_type(16))) __bf16 v16bf;
typedef __attribute__((ext_vector_type(8)))  float  v8f;

union FragBF { v16bf v; uint4 q[2]; };

#define EP_RELU       1
#define EP_ACCUM      2
#define EP_CLAMP_PM8  4
#define EP_CLAMP_MAX8 8
#define EP_TANH_TR    16

// ---------------------------------------------------------------------------
// Generic WMMA conv/GEMM:  Y[b, m, t] = sum_tap W_tap[m, :] . X[b, :, t+shift_tap]
// X viewed as channels [x_c0, x_c0+Cin) of a [B, x_ctot, T] bf16 tensor.
// Block: 128 threads = 4 waves; block tile M=128 x N=64; wave tile 64x32.
// ---------------------------------------------------------------------------
__global__ __launch_bounds__(128)
void wmma_conv(const __hip_bfloat16* __restrict__ X, int x_ctot, int x_c0,
               const __hip_bfloat16* __restrict__ W0,
               const __hip_bfloat16* __restrict__ W1,
               const float* __restrict__ bias,
               int Cin, int Cout, int shift0, int shift1,
               float* __restrict__ out_f32,
               __hip_bfloat16* __restrict__ out_bf16,
               int o_ctot, int o_c0, int flags)
{
  // X tile staged transposed: lds[t_local][cin] so K (channels) is contiguous.
  __shared__ __align__(16) __hip_bfloat16 lds[64 * (256 + 8)];
  const int CP   = Cin + 8;               // padded row stride (bank decorrelate)
  const int tid  = threadIdx.x;
  const int lane = tid & 31;
  const int wave = tid >> 5;
  const int wm   = wave & 1;              // M sub-block: rows wm*64
  const int wn   = wave >> 1;             // N sub-block: cols wn*32
  const int b    = blockIdx.y;
  const int t0   = blockIdx.x * 64;
  const int mblk = blockIdx.z * 128;
  const int half = lane >> 4;             // lane group 0/1
  const int l16  = lane & 15;

  v8f acc[4][2];
#pragma unroll
  for (int i = 0; i < 4; ++i)
#pragma unroll
    for (int j = 0; j < 2; ++j)
#pragma unroll
      for (int k = 0; k < 8; ++k) acc[i][j][k] = 0.0f;

  const int ntaps = (W1 != nullptr) ? 2 : 1;
  const int mbase = mblk + wm * 64;
  const bool mact = (mbase < Cout);       // wave-uniform

  for (int tap = 0; tap < ntaps; ++tap) {
    const int shift = tap ? shift1 : shift0;
    const __hip_bfloat16* W = tap ? W1 : W0;

    __syncthreads();
    for (int idx = tid; idx < 64 * Cin; idx += 128) {
      int tl = idx & 63;
      int c  = idx >> 6;
      int tg = t0 + tl + shift;
      __hip_bfloat16 v = __float2bfloat16(0.0f);
      if (tg >= 0 && tg < T_)
        v = X[((size_t)b * x_ctot + x_c0 + c) * (size_t)T_ + tg];
      lds[tl * CP + c] = v;
    }
    __syncthreads();

    if (tap == 0 && ntaps == 2) {
      // gfx1250 global_prefetch_b8: warm tap-1 tile while tap 0 computes
      int tl = tid & 63, c = tid >> 6;
      int tg = t0 + tl + shift1;
      if (tg >= 0 && tg < T_)
        __builtin_prefetch(&X[((size_t)b * x_ctot + x_c0 + c) * (size_t)T_ + tg], 0, 1);
    }

    if (mact) {
      for (int k0 = 0; k0 < Cin; k0 += 32) {
        // B fragments (32x16, K x N): lane holds column n = l16,
        // 16 contiguous K values starting at k0 + half*16.
        FragBF bfrag[2];
#pragma unroll
        for (int ni = 0; ni < 2; ++ni) {
          int tl = wn * 32 + ni * 16 + l16;
          const uint4* p = (const uint4*)&lds[tl * CP + k0 + half * 16];
          bfrag[ni].q[0] = p[0];
          bfrag[ni].q[1] = p[1];
        }
        // A fragments (16x32, M x K): lane holds row m = l16,
        // 16 contiguous K values starting at k0 + half*16.
        FragBF afrag[4];
#pragma unroll
        for (int mi = 0; mi < 4; ++mi) {
          int m = mbase + mi * 16 + l16;
          const __hip_bfloat16* wrow = W + (size_t)m * Cin;
          afrag[mi].q[0] = *(const uint4*)(wrow + k0 + half * 16);
          afrag[mi].q[1] = *(const uint4*)(wrow + k0 + half * 16 + 8);
        }
#pragma unroll
        for (int mi = 0; mi < 4; ++mi)
#pragma unroll
          for (int ni = 0; ni < 2; ++ni)
            acc[mi][ni] = __builtin_amdgcn_wmma_f32_16x16x32_bf16(
                false, afrag[mi].v, false, bfrag[ni].v,
                (short)0, acc[mi][ni], false, false);
      }
    }
  }

  if (!mact) return;
  // D layout (16x16 f32): lane holds n = l16; VGPR r holds m = half*8 + r.
#pragma unroll
  for (int mi = 0; mi < 4; ++mi) {
#pragma unroll
    for (int ni = 0; ni < 2; ++ni) {
      int t = t0 + wn * 32 + ni * 16 + l16;
#pragma unroll
      for (int r = 0; r < 8; ++r) {
        int m = mbase + mi * 16 + half * 8 + r;
        float v = acc[mi][ni][r];
        if (bias) v += bias[m];
        if (flags & EP_CLAMP_PM8)  v = fminf(fmaxf(v, -8.0f), 8.0f);
        if (flags & EP_CLAMP_MAX8) v = fminf(v, 8.0f);
        if (flags & EP_RELU)       v = fmaxf(v, 0.0f);
        if (flags & EP_TANH_TR) {
          // store [B, T, Cout] with tanh (output head)
          out_f32[((size_t)b * T_ + t) * o_ctot + o_c0 + m] = tanhf(v);
        } else {
          size_t oi = ((size_t)b * o_ctot + o_c0 + m) * (size_t)T_ + t;
          if (flags & EP_ACCUM) v += out_f32[oi];
          if (out_f32)  out_f32[oi]  = v;
          if (out_bf16) out_bf16[oi] = __float2bfloat16(v);
        }
      }
    }
  }
}

// --------------------------- support kernels -------------------------------
__global__ void cvt_weight(const float* __restrict__ w,
                           __hip_bfloat16* __restrict__ o,
                           int n, int K, int tap) {
  int i = blockIdx.x * 256 + threadIdx.x;
  if (i < n) o[i] = __float2bfloat16(w[(size_t)i * K + tap]);
}

__global__ void transpose_cvt(const float* __restrict__ in,
                              __hip_bfloat16* __restrict__ out, int C) {
  // in [B, T, C] f32 -> out [B, C, T] bf16
  size_t i = (size_t)blockIdx.x * 256 + threadIdx.x;
  size_t n = (size_t)B_ * T_ * C;
  if (i >= n) return;
  int c = (int)(i % C);
  size_t bt = i / C;
  int t = (int)(bt % T_);
  int b = (int)(bt / T_);
  out[((size_t)b * C + c) * (size_t)T_ + t] = __float2bfloat16(in[i]);
}

__global__ void gate_combine(const float* __restrict__ f,
                             const float* __restrict__ g,
                             const float* __restrict__ r,
                             __hip_bfloat16* __restrict__ d1, int ct1, int c01,
                             __hip_bfloat16* __restrict__ d2, int ct2, int c02) {
  size_t i = (size_t)blockIdx.x * 256 + threadIdx.x;
  if (i >= (size_t)B_ * D_ * T_) return;
  int t = (int)(i % T_);
  size_t bc = i / T_;
  int c = (int)(bc % D_);
  int b = (int)(bc / D_);
  float v = tanhf(f[i]) * (1.0f / (1.0f + expf(-g[i]))) + r[i];
  __hip_bfloat16 h = __float2bfloat16(v);
  d1[((size_t)b * ct1 + c01 + c) * (size_t)T_ + t] = h;
  if (d2) d2[((size_t)b * ct2 + c02 + c) * (size_t)T_ + t] = h;
}

__global__ void relu_cvt(const float* __restrict__ in,
                         __hip_bfloat16* __restrict__ out) {
  size_t i = (size_t)blockIdx.x * 256 + threadIdx.x;
  if (i < (size_t)B_ * D_ * T_) out[i] = __float2bfloat16(fmaxf(in[i], 0.0f));
}

__global__ void copy_ch(const __hip_bfloat16* __restrict__ src, int sct, int sc0,
                        __hip_bfloat16* __restrict__ dst, int dct, int dc0, int C) {
  size_t i = (size_t)blockIdx.x * 256 + threadIdx.x;
  if (i >= (size_t)B_ * C * T_) return;
  int t = (int)(i % T_);
  size_t bc = i / T_;
  int c = (int)(bc % C);
  int b = (int)(bc / C);
  dst[((size_t)b * dct + dc0 + c) * (size_t)T_ + t] =
      src[((size_t)b * sct + sc0 + c) * (size_t)T_ + t];
}

__global__ void zero_scalar(float* p) {
  if (threadIdx.x == 0 && blockIdx.x == 0) *p = 0.0f;
}

__device__ inline unsigned pcg(unsigned x) {
  x = x * 747796405u + 2891336453u;
  unsigned w = ((x >> ((x >> 28) + 4u)) ^ x) * 277803737u;
  return (w >> 22) ^ w;
}
__device__ inline float hnormal(unsigned idx, unsigned layer) {
  unsigned h1 = pcg(idx ^ (layer * 0x9E3779B9u));
  unsigned h2 = pcg(h1 + 0x85EBCA6Bu);
  float u1 = (float)(h1 >> 8) * (1.0f / 16777216.0f) + 1e-7f;
  float u2 = (float)(h2 >> 8) * (1.0f / 16777216.0f);
  return sqrtf(-2.0f * logf(u1)) * __cosf(6.28318530718f * u2);
}

__global__ __launch_bounds__(256)
void sampler_kld(const float* __restrict__ zp, const float* __restrict__ zq,
                 const float* __restrict__ mask,
                 __hip_bfloat16* __restrict__ catZ,
                 float* __restrict__ kld_out, int layer) {
  __shared__ float red[256];
  size_t i = (size_t)blockIdx.x * 256 + threadIdx.x;
  const size_t N = (size_t)B_ * D_ * T_;
  float kacc = 0.0f;
  if (i < N) {
    int t = (int)(i % T_);
    size_t bc = i / T_;
    int c = (int)(bc % D_);
    int b = (int)(bc / D_);
    size_t lo = ((size_t)b * 256 + c) * (size_t)T_ + t;
    size_t hi = ((size_t)b * 256 + 128 + c) * (size_t)T_ + t;
    float mu_p = zp[lo], th_p = zp[hi];
    float mu_q = zq[lo] + mu_p;           // Fraccaro trick
    float th_q = zq[hi];
    float eps  = hnormal((unsigned)i, (unsigned)layer);
    float z    = mu_q + eps * expf(0.5f * th_q);
    float dm   = mu_q - mu_p;
    float kld  = 0.5f * (th_p - th_q + expf(th_q - th_p) +
                         dm * dm * expf(-th_p) - 1.0f);
    kacc = kld * mask[(size_t)b * T_ + t] * (1.0f / (float)T_);
    catZ[hi] = __float2bfloat16(z);
  }
  red[threadIdx.x] = kacc;
  __syncthreads();
  for (int s = 128; s > 0; s >>= 1) {
    if (threadIdx.x < s) red[threadIdx.x] += red[threadIdx.x + s];
    __syncthreads();
  }
  if (threadIdx.x == 0) atomicAdd(kld_out, red[0]);
}

// ------------------------------- host ---------------------------------------
extern "C" void kernel_launch(void* const* d_in, const int* in_sizes, int n_in,
                              void* d_out, int out_size, void* d_ws, size_t ws_size,
                              hipStream_t stream) {
  (void)in_sizes; (void)n_in; (void)out_size; (void)ws_size;
  typedef __hip_bfloat16 bf;

  // ---- param leaf indices (jax pytree: dict keys sorted at every level) ----
  const int P = 3;
  const int I_BWD_GATES = P + 0;    // 4 x {b1,b2,w1,w2}
  const int I_BWD_WN    = P + 16;   // 4 x {bf,bg,br,wf,wg,wr}
  const int I_EMB_B     = P + 40;
  const int I_EMB_W     = P + 41;
  const int I_F1_B      = P + 42;
  const int I_F1_W      = P + 43;
  const int I_F2_B      = P + 44;
  const int I_F2_W      = P + 45;
  const int I_GATES_ST  = P + 46;   // 4 x gp
  const int I_INFER     = P + 62;   // 4 x gp
  const int I_INIT_SKIP = P + 78;   // 1 x gp
  const int I_OUT_B     = P + 82;
  const int I_OUT_W     = P + 83;
  const int I_PRIOR     = P + 84;   // 4 x gp
  const int I_SKIP_NS   = P + 100;  // 6 x gp
  const int I_SKIP_ST   = P + 124;  // 4 x gp
  const int I_WN_NS     = P + 140;  // 6 x wp
  const int I_WN_ST     = P + 176;  // 4 x wp

  // ---- workspace bump allocator ----
  size_t ws_off = 0;
  auto wsAlloc = [&](size_t bytes) -> void* {
    void* p = (char*)d_ws + ws_off;
    ws_off += (bytes + 255) & ~(size_t)255;
    return p;
  };
  const size_t A1  = (size_t)B_ * D_ * T_;       // elements, 128-ch tensor
  const size_t A2  = (size_t)B_ * 256 * T_;      // elements, 256-ch tensor
  const size_t AIN = (size_t)B_ * DIN_ * T_;

  bf* xbf   = (bf*)wsAlloc(AIN * 2);
  bf* ybf   = (bf*)wsAlloc(AIN * 2);
  bf* catY  = (bf*)wsAlloc(A2 * 2);              // [xb | target]
  bf* catB0 = (bf*)wsAlloc(A2 * 2);              // [nx | backwards[i]]
  bf* catB1 = (bf*)wsAlloc(A2 * 2);
  bf* catB2 = (bf*)wsAlloc(A2 * 2);
  bf* catB3 = (bf*)wsAlloc(A2 * 2);
  bf* catBl[4] = {catB0, catB1, catB2, catB3};
  bf* catZ  = (bf*)wsAlloc(A2 * 2);              // [nx | z]
  bf* xfA   = (bf*)wsAlloc(A1 * 2);
  bf* xfB   = (bf*)wsAlloc(A1 * 2);
  bf* g1    = (bf*)wsAlloc(A1 * 2);              // gates() hidden (relu)
  float* tmpF   = (float*)wsAlloc(A1 * 4);
  float* tmpG   = (float*)wsAlloc(A1 * 4);
  float* tmpR   = (float*)wsAlloc(A1 * 4);
  float* finalB = (float*)wsAlloc(A1 * 4);
  float* zp     = (float*)wsAlloc(A2 * 4);
  float* zq     = (float*)wsAlloc(A2 * 4);

  float* outp   = (float*)d_out;                 // [B, T, DOUT] then kld
  float* kldp   = outp + (size_t)B_ * T_ * DOUT_;

  auto fleaf = [&](int i) { return (const float*)d_in[i]; };

  auto cvtW = [&](int leaf, int O, int I, int K, int tap) -> bf* {
    bf* dst = (bf*)wsAlloc((size_t)O * I * 2);
    int n = O * I;
    cvt_weight<<<dim3((n + 255) / 256), dim3(256), 0, stream>>>(
        fleaf(leaf), dst, n, K, tap);
    return dst;
  };

  auto conv = [&](const bf* X, int xct, int xc0, const bf* W0, const bf* W1,
                  const float* bias, int Cin, int Cout, int s0, int s1,
                  float* of, bf* ob, int oct, int oc0, int fl) {
    dim3 grid(T_ / 64, B_, (Cout + 127) / 128);
    wmma_conv<<<grid, dim3(128), 0, stream>>>(X, xct, xc0, W0, W1, bias,
                                              Cin, Cout, s0, s1,
                                              of, ob, oct, oc0, fl);
  };

  // gates(x) = conv2(relu(conv1(x))) ; gp leaves: b1,b2,w1,w2 ; hidden = D
  auto gates_op = [&](const bf* X, int xct, int Cin, int base, int Cout,
                      float* of, bf* ob, int oct, int oc0, int fl) {
    bf* w1 = cvtW(base + 2, D_, Cin, 1, 0);
    bf* w2 = cvtW(base + 3, Cout, D_, 1, 0);
    conv(X, xct, 0, w1, nullptr, fleaf(base + 0), Cin, D_, 0, 0,
         nullptr, g1, D_, 0, EP_RELU);
    conv(g1, D_, 0, w2, nullptr, fleaf(base + 1), D_, Cout, 0, 0,
         of, ob, oct, oc0, fl);
  };

  const size_t NG = (A1 + 255) / 256;
  // wavenet_gate: tanh(conv_f)*sigmoid(conv_g) + conv_r ; wp: bf,bg,br,wf,wg,wr
  auto wgate = [&](const bf* X, int xct, int xc0, int base, int dil, bool fwd,
                   bf* d1, int ct1, int c01, bf* d2, int ct2, int c02) {
    bf* wf0 = cvtW(base + 3, D_, D_, 2, 0);
    bf* wf1 = cvtW(base + 3, D_, D_, 2, 1);
    bf* wg0 = cvtW(base + 4, D_, D_, 2, 0);
    bf* wg1 = cvtW(base + 4, D_, D_, 2, 1);
    bf* wr0 = cvtW(base + 5, D_, D_, 1, 0);
    int s0 = fwd ? -dil : 0;
    int s1 = fwd ? 0 : dil;
    conv(X, xct, xc0, wf0, wf1, fleaf(base + 0), D_, D_, s0, s1,
         tmpF, nullptr, D_, 0, 0);
    conv(X, xct, xc0, wg0, wg1, fleaf(base + 1), D_, D_, s0, s1,
         tmpG, nullptr, D_, 0, 0);
    conv(X, xct, xc0, wr0, nullptr, fleaf(base + 2), D_, D_, 0, 0,
         tmpR, nullptr, D_, 0, 0);
    gate_combine<<<dim3((unsigned)NG), dim3(256), 0, stream>>>(
        tmpF, tmpG, tmpR, d1, ct1, c01, d2, ct2, c02);
  };

  // ---------------------------- pipeline ----------------------------------
  zero_scalar<<<1, 64, 0, stream>>>(kldp);

  { // transpose + bf16 inputs
    size_t n = AIN;
    transpose_cvt<<<dim3((unsigned)((n + 255) / 256)), dim3(256), 0, stream>>>(
        (const float*)d_in[0], xbf, DIN_);
    transpose_cvt<<<dim3((unsigned)((n + 255) / 256)), dim3(256), 0, stream>>>(
        (const float*)d_in[1], ybf, DIN_);
  }
  const float* mask = (const float*)d_in[2];

  bf* emb_w = cvtW(I_EMB_W, D_, DIN_, 1, 0);
  const float* emb_b = fleaf(I_EMB_B);

  // ---- backward (anti-causal) pass over y ----
  conv(ybf, DIN_, 0, emb_w, nullptr, emb_b, DIN_, D_, 0, 0,
       nullptr, catY, 256, 0, 0);                         // emb_y -> xb slot
  copy_ch<<<dim3((unsigned)NG), dim3(256), 0, stream>>>(catY, 256, 0,
                                                        catY, 256, 128, D_);
  {
    int dil = 512;
    for (int i = 0; i < NSL_; ++i) {
      wgate(catY, 256, 0, I_BWD_WN + (NSL_ - 1 - i) * 6, dil, /*fwd=*/false,
            catY, 256, 0, nullptr, 0, 0);                 // xb in-place
      gates_op(catY, 256, 256, I_BWD_GATES + i * 4, D_,
               nullptr, catBl[i], 256, 128, 0);           // backwards[i] -> bv
      if (dil > 1) dil /= 2;
    }
  }

  // ---- causal forward stack ----
  conv(xbf, DIN_, 0, emb_w, nullptr, emb_b, DIN_, D_, 0, 0,
       nullptr, xfA, D_, 0, EP_RELU);                     // xf = relu(emb(x))
  gates_op(xfA, D_, D_, I_INIT_SKIP, D_, finalB, nullptr, D_, 0, 0);

  bf* xf = xfA;
  bf* xo = xfB;
  int dil = 1;
  for (int i = 0; i < 6; ++i) {                           // NUM_LAYERS - NSL
    wgate(xf, D_, 0, I_WN_NS + i * 6, dil, true, xo, D_, 0, nullptr, 0, 0);
    gates_op(xo, D_, D_, I_SKIP_NS + i * 4, D_, finalB, nullptr, D_, 0, EP_ACCUM);
    bf* t = xf; xf = xo; xo = t;
    dil *= 2;
  }

  // ---- stochastic layers ----
  for (int i = 0; i < NSL_; ++i) {
    bf* cb = catBl[NSL_ - 1 - i];                         // [nx | bv]
    wgate(xf, D_, 0, I_WN_ST + i * 6, dil, true,
          cb, 256, 0, catZ, 256, 0);                      // nx into both cats
    gates_op(cb, 256, D_, I_PRIOR + i * 4, 2 * D_,
             zp, nullptr, 256, 0, EP_CLAMP_PM8);          // zp = clip(prior(nx))
    gates_op(cb, 256, 256, I_INFER + i * 4, 2 * D_,
             zq, nullptr, 256, 0, EP_CLAMP_MAX8);         // zq = min(inf(cat),8)
    sampler_kld<<<dim3((unsigned)NG), dim3(256), 0, stream>>>(
        zp, zq, mask, catZ, kldp, i);                     // z + KLD
    gates_op(catZ, 256, 256, I_SKIP_ST + i * 4, D_,
             finalB, nullptr, D_, 0, EP_ACCUM);           // final += skip
    gates_op(catZ, 256, 256, I_GATES_ST + i * 4, D_,
             nullptr, xo, D_, 0, 0);                      // xf = gates_st(cat)
    bf* t = xf; xf = xo; xo = t;
    dil *= 2;
  }

  // ---- output head ----
  relu_cvt<<<dim3((unsigned)NG), dim3(256), 0, stream>>>(finalB, xo);
  bf* f1w = cvtW(I_F1_W, D_, D_, 1, 0);
  bf* f2w = cvtW(I_F2_W, D_, D_, 1, 0);
  bf* oww = cvtW(I_OUT_W, DOUT_, D_, 1, 0);
  conv(xo, D_, 0, f1w, nullptr, fleaf(I_F1_B), D_, D_, 0, 0,
       nullptr, xf, D_, 0, EP_RELU);                      // relu(f1(.)) bf16
  conv(xf, D_, 0, f2w, nullptr, fleaf(I_F2_B), D_, D_, 0, 0,
       nullptr, g1, D_, 0, 0);                            // f2 output bf16
  conv(g1, D_, 0, oww, nullptr, fleaf(I_OUT_B), D_, DOUT_, 0, 0,
       outp, nullptr, DOUT_, 0, EP_TANH_TR);              // tanh + [B,T,DOUT]
}
